// TextGuidedVisionPruner_75634374082783
// MI455X (gfx1250) — matrix operand order repaired and checked by reading, not compiled, not run on hardware
//
#include <hip/hip_runtime.h>
#include <hip/hip_bf16.h>
#include <math.h>

// ---------------- problem constants (from reference) ----------------
#define Bc   64
#define LVc  1024
#define LTc  128
#define Dc   256
#define TDc  1536
#define Hc   4
#define HDc  64
#define Kc   512      // kept tokens
#define NDc  512      // dropped tokens
#define EPSN 1e-12f

typedef __attribute__((ext_vector_type(16))) _Float16 v16h;
typedef __attribute__((ext_vector_type(8)))  _Float16 v8h;
typedef __attribute__((ext_vector_type(8)))  float    v8f;

// ---------------- WMMA fragment loaders (wave32, 16x16x32 f16) ------
// A-matrix 16x32 f16 layout: lane l holds row m=l%16;
// lanes 0-15: K=0..15; lanes 16-31: K=16..31 (matches B fragment).
// f16 source: two aligned 16-byte vector loads, spliced register-only.
__device__ inline v16h load_a_frag_h(const _Float16* __restrict__ src, int lda,
                                     int m0, int k0, int lane) {
    int h = lane >> 4, m = lane & 15;
    const _Float16* row = src + (size_t)(m0 + m) * lda + k0 + 16 * h;
    v8h lo = *(const v8h*)row;          // K = k0+16h   .. +7
    v8h hi = *(const v8h*)(row + 8);    // K = k0+16h+8 .. +15
    return __builtin_shufflevector(lo, hi, 0, 1, 2, 3, 4, 5, 6, 7,
                                   8, 9, 10, 11, 12, 13, 14, 15);
}

// B-matrix 32x16 f16 (K x N), B[k][n] = W[n0+n][k0+k] from row-major f16 W[N][ldw].
// lane l holds column n=l%16; lanes 0-15: K=0..15, lanes 16-31: K=16..31.
// One aligned 32-byte vector load.
__device__ inline v16h load_b_frag_h(const _Float16* __restrict__ W, int ldw,
                                     int n0, int k0, int lane) {
    int h = lane >> 4, n = lane & 15;
    return *(const v16h*)(W + (size_t)(n0 + n) * ldw + k0 + 16 * h);
}

__device__ inline float block_reduce_sum(float v, float* sdata) {
    int t = threadIdx.x;
    sdata[t] = v;
    __syncthreads();
    for (int s = blockDim.x >> 1; s > 0; s >>= 1) {
        if (t < s) sdata[t] += sdata[t + s];
        __syncthreads();
    }
    float r = sdata[0];
    __syncthreads();
    return r;
}

// ---------------- f32 -> f16 bulk convert ----------------------------
__global__ void cvt_f16_kernel(const float* __restrict__ src,
                               _Float16* __restrict__ dst, size_t n) {
    size_t i = ((size_t)blockIdx.x * blockDim.x + threadIdx.x) * 4;
    if (i + 3 < n) {
        float4 v = *(const float4*)(src + i);
        dst[i + 0] = (_Float16)v.x;
        dst[i + 1] = (_Float16)v.y;
        dst[i + 2] = (_Float16)v.z;
        dst[i + 3] = (_Float16)v.w;
    }
}

// ---------------- GEMM: C[M,N] = A[M,Kd] * W[N,Kd]^T + bias[N] -------
// f16 operands, f32 accumulate, f16 result.
// one wave = 16 rows x 64 cols (A fragment reused across 4 N-tiles)
__global__ void gemm_bias_h_kernel(const _Float16* __restrict__ A,
                                   const _Float16* __restrict__ W,
                                   const float* __restrict__ bias,
                                   _Float16* __restrict__ C,
                                   int M, int N, int Kd) {
    int lane = threadIdx.x & 31;
    int wave = threadIdx.x >> 5;
    int wpb  = blockDim.x >> 5;
    int tiles_n = N / 64;
    int wid = blockIdx.x * wpb + wave;
    int m0 = (wid / tiles_n) * 16;
    int n0 = (wid % tiles_n) * 64;
    if (m0 >= M) return;

    v8f acc[4] = {};
    for (int k0 = 0; k0 < Kd; k0 += 32) {
        v16h a = load_a_frag_h(A, Kd, m0, k0, lane);
        if (k0 + 32 < Kd)   // stream-prefetch next A k-tile (global_prefetch_b8)
            __builtin_prefetch(A + (size_t)(m0 + (lane & 15)) * Kd + k0 + 32, 0, 1);
#pragma unroll
        for (int t = 0; t < 4; ++t) {
            v16h b = load_b_frag_h(W, Kd, n0 + 16 * t, k0, lane);
            acc[t] = __builtin_amdgcn_wmma_f32_16x16x32_f16(
                false, a, false, b, (short)0, acc[t], false, false);
        }
    }
    int h = lane >> 4, n = lane & 15;
#pragma unroll
    for (int t = 0; t < 4; ++t)
#pragma unroll
        for (int r = 0; r < 8; ++r) {
            int row = m0 + r + 8 * h, col = n0 + 16 * t + n;
            C[(size_t)row * N + col] = (_Float16)(acc[t][r] + bias[col]);
        }
}

// ---------------- sentiment weights w_t[b][k] ------------------------
__global__ void wt_kernel(const float* __restrict__ senti,
                          const unsigned char* __restrict__ mt,
                          float* __restrict__ wt) {
    __shared__ float red[LTc];
    int b = blockIdx.x, t = threadIdx.x;   // 128 threads
    float m = mt[b * LTc + t] ? 1.f : 0.f;
    float w = fabsf(senti[b * LTc + t]) * m;
    red[t] = w;
    __syncthreads();
    for (int s = LTc / 2; s > 0; s >>= 1) {
        if (t < s) red[t] += red[t + s];
        __syncthreads();
    }
    wt[b * LTc + t] = w / (red[0] + 1e-8f) * m;
}

// ---------------- attention scores -> score_v ------------------------
// block = (batch b, 16 query rows); 4 waves, wave = head. f16 Q/K operands.
__global__ void attn_score_kernel(const _Float16* __restrict__ Q,
                                  const _Float16* __restrict__ Km,
                                  const float* __restrict__ wt,
                                  const unsigned char* __restrict__ mask_t,
                                  const unsigned char* __restrict__ mask_v,
                                  float* __restrict__ score_v) {
    __shared__ float s_sc[Hc * 16 * LTc];   // 32 KB score tile
    __shared__ float s_red[16 * Hc];
    int blk = blockIdx.x;
    int b  = blk / (LVc / 16);
    int q0 = (blk % (LVc / 16)) * 16;
    int lane = threadIdx.x & 31;
    int head = threadIdx.x >> 5;            // 0..3

    const _Float16* Qp = Q  + ((size_t)b * LVc + q0) * Dc;
    const _Float16* Kp = Km + (size_t)b * LTc * Dc;

    v8f acc[LTc / 16] = {};
    for (int k0 = 0; k0 < HDc; k0 += 32) {
        v16h a = load_a_frag_h(Qp, Dc, 0, head * HDc + k0, lane);
#pragma unroll
        for (int t = 0; t < LTc / 16; ++t) {
            v16h bf = load_b_frag_h(Kp, Dc, t * 16, head * HDc + k0, lane);
            acc[t] = __builtin_amdgcn_wmma_f32_16x16x32_f16(
                false, a, false, bf, (short)0, acc[t], false, false);
        }
    }
    int h = lane >> 4, n = lane & 15;
#pragma unroll
    for (int t = 0; t < LTc / 16; ++t)
#pragma unroll
        for (int r = 0; r < 8; ++r) {
            int row = r + 8 * h, col = t * 16 + n;
            float v = acc[t][r] * 0.125f;          // 1/sqrt(HD)
            if (!mask_t[b * LTc + col]) v = -1e9f;
            s_sc[(head * 16 + row) * LTc + col] = v;
        }
    __syncthreads();

    int tt = threadIdx.x;
    if (tt < 64) {                 // (head hh, row r) softmax + dot w_t
        int hh = tt >> 4, r = tt & 15;
        const float* rowp = s_sc + (hh * 16 + r) * LTc;
        float m = -3e38f;
        for (int c = 0; c < LTc; ++c) m = fmaxf(m, rowp[c]);
        float sum = 0.f, acw = 0.f;
        for (int c = 0; c < LTc; ++c) {
            float e = __expf(rowp[c] - m);
            sum += e;
            acw += e * wt[b * LTc + c];
        }
        s_red[r * Hc + hh] = acw / sum;
    }
    __syncthreads();
    if (tt < 16) {
        float s = 0.f;
        for (int hh = 0; hh < Hc; ++hh) s += s_red[tt * Hc + hh];
        s *= (1.0f / Hc);
        if (!mask_v[b * LVc + q0 + tt]) s = -1e9f;
        score_v[b * LVc + q0 + tt] = s;
    }
}

// ---------------- top-K selection (rank by counting) -----------------
__global__ void topk_kernel(const float* __restrict__ score_v,
                            int* __restrict__ keepI, int* __restrict__ dropI,
                            int* __restrict__ out_idx) {
    __shared__ float sc[LVc];
    __shared__ int   kept[LVc];
    int b = blockIdx.x, t = threadIdx.x;
    for (int i = t; i < LVc; i += blockDim.x) sc[i] = score_v[b * LVc + i];
    __syncthreads();
    for (int i = t; i < LVc; i += blockDim.x) {
        float si = sc[i];
        int rank = 0;
        for (int j = 0; j < LVc; ++j) {
            float sj = sc[j];
            rank += (sj > si) || (sj == si && j < i);   // stable descending rank
        }
        kept[i] = (rank < Kc) ? 1 : 0;
    }
    __syncthreads();
    for (int i = t; i < LVc; i += blockDim.x) {
        int pos = 0;
        if (kept[i]) {
            for (int j = 0; j < i; ++j) pos += kept[j];
            keepI[b * Kc + pos]   = i;
            out_idx[b * Kc + pos] = i;
        } else {
            for (int j = 0; j < i; ++j) pos += 1 - kept[j];
            dropI[b * NDc + pos] = i;
        }
    }
}

// ---------------- gather + l2-normalize kept/dropped rows (f16 out) --
__global__ void gather_norm_kernel(const float* __restrict__ hv,
                                   const int* __restrict__ keepI,
                                   const int* __restrict__ dropI,
                                   const float* __restrict__ score_v,
                                   _Float16* __restrict__ vkn,
                                   _Float16* __restrict__ vdn,
                                   float* __restrict__ sd) {
    __shared__ float red[Dc];
    int blk = blockIdx.x;
    int b = blk / LVc, r = blk % LVc;
    int c = threadIdx.x;
    int src;
    _Float16* dst;
    if (r < Kc) {
        src = keepI[b * Kc + r];
        dst = vkn + ((size_t)b * Kc + r) * Dc;
    } else {
        int j = r - Kc;
        src = dropI[b * NDc + j];
        dst = vdn + ((size_t)b * NDc + j) * Dc;
        if (c == 0) sd[b * NDc + j] = score_v[b * LVc + src];
    }
    float v  = hv[((size_t)b * LVc + src) * Dc + c];
    float ss = block_reduce_sum(v * v, red);
    dst[c] = (_Float16)(v / fmaxf(sqrtf(ss), EPSN));
}

// ---------------- sim = Vd_n @ Vk_n^T ; per-row argmax ---------------
// block = (batch b, 16 dropped rows); 8 waves x 64 kept-cols each.
__global__ void sim_assign_kernel(const _Float16* __restrict__ vdn,
                                  const _Float16* __restrict__ vkn,
                                  int* __restrict__ assignI) {
    __shared__ float s_sim[16 * Kc];      // 32 KB
    int blk = blockIdx.x;
    int b  = blk / (NDc / 16);
    int d0 = (blk % (NDc / 16)) * 16;
    int lane = threadIdx.x & 31;
    int wave = threadIdx.x >> 5;          // 0..7
    const _Float16* Ap = vdn + ((size_t)b * NDc + d0) * Dc;
    const _Float16* Bp = vkn + (size_t)b * Kc * Dc;

    v8f acc[4] = {};
    int nbase = wave * 64;
    for (int k0 = 0; k0 < Dc; k0 += 32) {
        v16h a = load_a_frag_h(Ap, Dc, 0, k0, lane);
#pragma unroll
        for (int t = 0; t < 4; ++t) {
            v16h bf = load_b_frag_h(Bp, Dc, nbase + 16 * t, k0, lane);
            acc[t] = __builtin_amdgcn_wmma_f32_16x16x32_f16(
                false, a, false, bf, (short)0, acc[t], false, false);
        }
    }
    int h = lane >> 4, n = lane & 15;
#pragma unroll
    for (int t = 0; t < 4; ++t)
#pragma unroll
        for (int r = 0; r < 8; ++r)
            s_sim[(r + 8 * h) * Kc + nbase + 16 * t + n] = acc[t][r];
    __syncthreads();

    if (threadIdx.x < 16) {
        int r = threadIdx.x;
        const float* rp = s_sim + r * Kc;
        float best = rp[0];
        int bi = 0;
        for (int c = 1; c < Kc; ++c)
            if (rp[c] > best) { best = rp[c]; bi = c; }   // first max (jnp.argmax)
        assignI[b * NDc + d0 + r] = bi;
    }
}

// ---------------- segment max / denom / member count -----------------
__global__ void seg_stats_kernel(const float* __restrict__ sd,
                                 const int* __restrict__ assignI,
                                 float* __restrict__ smax,
                                 float* __restrict__ denom,
                                 int* __restrict__ cnt) {
    int b = blockIdx.x;
    int j = threadIdx.x;                  // cluster id, 512 threads
    float m = -3.0e38f;
    int n = 0;
    for (int d = 0; d < NDc; ++d)
        if (assignI[b * NDc + d] == j) { m = fmaxf(m, sd[b * NDc + d]); ++n; }
    float den = 0.f;
    for (int d = 0; d < NDc; ++d)
        if (assignI[b * NDc + d] == j) den += __expf(sd[b * NDc + d] - m);
    smax[b * Kc + j]  = m;
    denom[b * Kc + j] = den;
    cnt[b * Kc + j]   = n;
}

// ---------------- scatter merged = segsum(w * V_drop) ----------------
__global__ void merge_scatter_kernel(const float* __restrict__ hv,
                                     const int* __restrict__ dropI,
                                     const int* __restrict__ assignI,
                                     const float* __restrict__ sd,
                                     const float* __restrict__ smax,
                                     const float* __restrict__ denom,
                                     float* __restrict__ merged) {
    int blk = blockIdx.x;
    int b = blk / NDc, d = blk % NDc;
    int a = assignI[b * NDc + d];
    float w = __expf(sd[b * NDc + d] - smax[b * Kc + a]) /
              fmaxf(denom[b * Kc + a], EPSN);
    int src = dropI[b * NDc + d];
    int c = threadIdx.x;
    atomicAdd(&merged[((size_t)b * Kc + a) * Dc + c],
              w * hv[((size_t)b * LVc + src) * Dc + c]);
}

// ---------------- finalize: l2norm(V_keep+merged) / passthrough ------
__global__ void finalize_kernel(const float* __restrict__ hv,
                                const int* __restrict__ keepI,
                                const float* __restrict__ merged,
                                const int* __restrict__ cnt,
                                float* __restrict__ out) {
    __shared__ float red[Dc];
    int blk = blockIdx.x;
    int b = blk / Kc, j = blk % Kc;
    int c = threadIdx.x;
    int src = keepI[b * Kc + j];
    float vk  = hv[((size_t)b * LVc + src) * Dc + c];
    float val = vk + merged[((size_t)b * Kc + j) * Dc + c];
    float ss  = block_reduce_sum(val * val, red);
    float o   = (cnt[b * Kc + j] > 0) ? (val / fmaxf(sqrtf(ss), EPSN)) : vk;
    out[((size_t)b * Kc + j) * Dc + c] = o;
}

__global__ void fill_zero_kernel(float* __restrict__ p, size_t n) {
    size_t i = (size_t)blockIdx.x * blockDim.x + threadIdx.x;
    if (i < n) p[i] = 0.f;
}

// ---------------------------------------------------------------------
extern "C" void kernel_launch(void* const* d_in, const int* in_sizes, int n_in,
                              void* d_out, int out_size, void* d_ws, size_t ws_size,
                              hipStream_t stream) {
    const float*         hidden_v = (const float*)d_in[0];
    const float*         hidden_t = (const float*)d_in[1];
    const float*         senti_t  = (const float*)d_in[2];
    const unsigned char* mask_v   = (const unsigned char*)d_in[3];
    const unsigned char* mask_t   = (const unsigned char*)d_in[4];
    const float*         W_t      = (const float*)d_in[5];
    const float*         b_t      = (const float*)d_in[6];
    const float*         Wq       = (const float*)d_in[7];
    const float*         bq       = (const float*)d_in[8];
    const float*         Wk       = (const float*)d_in[9];
    const float*         bk       = (const float*)d_in[10];

    float* out_f   = (float*)d_out;
    int*   out_idx = (int*)(out_f + (size_t)Bc * Kc * Dc);

    // workspace carve-up, 64B-aligned regions
    char*  base = (char*)d_ws;
    size_t off  = 0;
    auto alloc = [&](size_t bytes) -> void* {
        void* p = base + off;
        off = (off + bytes + 63) & ~(size_t)63;
        return p;
    };
    // f16 operand copies
    _Float16* hth  = (_Float16*)alloc((size_t)Bc * LTc * TDc * 2);  // hidden_t f16
    _Float16* hvh  = (_Float16*)alloc((size_t)Bc * LVc * Dc  * 2);  // hidden_v f16
    _Float16* Wth  = (_Float16*)alloc((size_t)Dc * TDc * 2);
    _Float16* Wqh  = (_Float16*)alloc((size_t)Dc * Dc  * 2);
    _Float16* Wkh  = (_Float16*)alloc((size_t)Dc * Dc  * 2);
    // f16 intermediates
    _Float16* Th   = (_Float16*)alloc((size_t)Bc * LTc * Dc * 2);
    _Float16* Kh   = (_Float16*)alloc((size_t)Bc * LTc * Dc * 2);
    _Float16* Qh   = (_Float16*)alloc((size_t)Bc * LVc * Dc * 2);
    _Float16* vkn  = (_Float16*)alloc((size_t)Bc * Kc  * Dc * 2);
    _Float16* vdn  = (_Float16*)alloc((size_t)Bc * NDc * Dc * 2);
    // f32 / int scratch
    float* wtws    = (float*)alloc((size_t)Bc * LTc * 4);
    float* svws    = (float*)alloc((size_t)Bc * LVc * 4);
    int*   keepI   = (int*)  alloc((size_t)Bc * Kc  * 4);
    int*   dropI   = (int*)  alloc((size_t)Bc * NDc * 4);
    float* sdws    = (float*)alloc((size_t)Bc * NDc * 4);
    int*   assignI = (int*)  alloc((size_t)Bc * NDc * 4);
    float* smax    = (float*)alloc((size_t)Bc * Kc  * 4);
    float* denom   = (float*)alloc((size_t)Bc * Kc  * 4);
    int*   cnt     = (int*)  alloc((size_t)Bc * Kc  * 4);
    float* merged  = (float*)alloc((size_t)Bc * Kc * Dc * 4);

    // 0) bulk f32 -> f16 conversions (one pass each; removes cvt from GEMM loops)
    auto cvt = [&](const float* s, _Float16* d, size_t n) {
        unsigned g = (unsigned)((n / 4 + 255) / 256);
        cvt_f16_kernel<<<g, 256, 0, stream>>>(s, d, n);
    };
    cvt(hidden_t, hth, (size_t)Bc * LTc * TDc);
    cvt(hidden_v, hvh, (size_t)Bc * LVc * Dc);
    cvt(W_t, Wth, (size_t)Dc * TDc);
    cvt(Wq,  Wqh, (size_t)Dc * Dc);
    cvt(Wk,  Wkh, (size_t)Dc * Dc);

    // 1) sentiment weights
    wt_kernel<<<Bc, LTc, 0, stream>>>(senti_t, mask_t, wtws);

    // 2) T = hidden_t @ W_t^T + b_t          [8192 x 256, K=1536]
    {
        int M = Bc * LTc, waves = (M / 16) * (Dc / 64);
        gemm_bias_h_kernel<<<waves / 8, 256, 0, stream>>>(hth, Wth, b_t, Th, M, Dc, TDc);
    }
    // 3) Kmat = T @ Wk^T + bk                [8192 x 256, K=256]
    {
        int M = Bc * LTc, waves = (M / 16) * (Dc / 64);
        gemm_bias_h_kernel<<<waves / 8, 256, 0, stream>>>(Th, Wkh, bk, Kh, M, Dc, Dc);
    }
    // 4) Q = hidden_v @ Wq^T + bq            [65536 x 256, K=256]
    {
        int M = Bc * LVc, waves = (M / 16) * (Dc / 64);
        gemm_bias_h_kernel<<<waves / 8, 256, 0, stream>>>(hvh, Wqh, bq, Qh, M, Dc, Dc);
    }
    // 5) per-head attention -> score_v
    attn_score_kernel<<<Bc * (LVc / 16), 128, 0, stream>>>(Qh, Kh, wtws, mask_t, mask_v, svws);
    // 6) top-K (sorted kept & dropped indices)
    topk_kernel<<<Bc, 256, 0, stream>>>(svws, keepI, dropI, out_idx);
    // 7) gather + normalize rows for cosine sim (f16 out)
    gather_norm_kernel<<<Bc * LVc, Dc, 0, stream>>>(hidden_v, keepI, dropI, svws, vkn, vdn, sdws);
    // 8) zero merged accumulators
    {
        size_t n = (size_t)Bc * Kc * Dc;
        fill_zero_kernel<<<(unsigned)((n + 255) / 256), 256, 0, stream>>>(merged, n);
    }
    // 9) cosine-sim GEMM + argmax assignment
    sim_assign_kernel<<<Bc * (NDc / 16), 256, 0, stream>>>(vdn, vkn, assignI);
    // 10) segment softmax stats
    seg_stats_kernel<<<Bc, Kc, 0, stream>>>(sdws, assignI, smax, denom, cnt);
    // 11) scatter weighted dropped rows into clusters
    merge_scatter_kernel<<<Bc * NDc, Dc, 0, stream>>>(hidden_v, dropI, assignI, sdws, smax, denom, merged);
    // 12) finalize output rows
    finalize_kernel<<<Bc * Kc, Dc, 0, stream>>>(hidden_v, keepI, merged, cnt, out_f);
}
